// MelPerformer_2113123909975
// MI455X (gfx1250) — compile-verified, matrix operand order checked by reading, not run on hardware
//
#include <hip/hip_runtime.h>
#include <math.h>

// ---------------- model constants ----------------
#define BB    8
#define TT    2048
#define NMEL  64
#define DD    256
#define HH    4
#define DHH   64
#define MFF   512
#define LLAY  4
#define BT    (BB * TT)          // 16384 rows
#define BHT   (BB * HH * TT)     // 65536 (b,h,t) rows

typedef _Float16 f16;
typedef __attribute__((ext_vector_type(16))) _Float16 v16h;
typedef __attribute__((ext_vector_type(8)))  float    v8f;

// CDNA5 async global->LDS staging if the toolchain exposes the builtins.
#if defined(__has_builtin)
#  if __has_builtin(__builtin_amdgcn_global_load_async_to_lds_b128) && \
      __has_builtin(__builtin_amdgcn_s_wait_asynccnt)
#    define USE_ASYNC_LDS 1
#  endif
#endif
#ifndef USE_ASYNC_LDS
#  define USE_ASYNC_LDS 0
#endif

#if USE_ASYNC_LDS
// Builtin expects: (v4i addrspace(1)*, v4i addrspace(3)*, imm offset, imm cpol)
typedef int vi4 __attribute__((vector_size(16)));
typedef __attribute__((address_space(1))) vi4 g_vi4;
typedef __attribute__((address_space(3))) vi4 l_vi4;
#endif

// =====================================================================
// Batched WMMA GEMM:  C[M,N] = op(A)[M,K] * op(B)[K,N]  (+bias) (+=C)
//   tA=0: A stored [M x K].  tA=1: A stored [K x M].
//   tB=0: B stored [K x N].  tB=1: B stored [N x K].
// Block = 128 threads (4 waves) -> 64x64 C tile; each wave 2x2 of
// 16x16 accumulators via v_wmma_f32_16x16x32_f16.
// LDS tiles are staged f32 in the SAME orientation as global memory, so
// the global->LDS copy is always contiguous 16B/lane segments (async
// tensor path / coalesced fallback); transposition is absorbed by the
// LDS->fragment gather, and f32->f16 conversion happens in registers.
// =====================================================================
struct GemmP {
    const float* A; const float* B; float* C; const float* bias;
    int M, N, K, lda, ldb, ldc;
    long long sA, sB, sC;     // batch strides (elements)
    int tA, tB, accum;
};

__global__ __launch_bounds__(128) void wmma_gemm_kernel(GemmP p) {
    const float* Ab = p.A + (long long)blockIdx.z * p.sA;
    const float* Bb = p.B + (long long)blockIdx.z * p.sB;
    float*       Cb = p.C + (long long)blockIdx.z * p.sC;
    const int rowBase = blockIdx.y * 64;
    const int colBase = blockIdx.x * 64;

    // Staging layouts (f32, mirror global orientation of the tile):
    //   tA=0: [64 m][stride 36 over k]    tA=1: [32 k][stride 68 over m]
    //   tB=0: [32 k][stride 68 over n]    tB=1: [64 n][stride 36 over k]
    __shared__ __align__(16) float AsBuf[64 * 36];
    __shared__ __align__(16) float BsBuf[64 * 36];

    const int tid  = threadIdx.x;
    const int lane = tid & 31;
    const int wv   = tid >> 5;
    const int wr   = (wv >> 1) * 32;     // wave row offset in tile
    const int wc   = (wv & 1) * 32;      // wave col offset in tile
    const int mloc = lane & 15;
    const int hsel = lane >> 4;          // lane-half selects K group

    v8f acc[2][2] = {};

    for (int k0 = 0; k0 < p.K; k0 += 32) {
        // ---- stage both tiles: 512 x 16B segments each, 4+4 per thread ----
        #pragma unroll
        for (int j = 0; j < 4; ++j) {
            const int g = tid * 4 + j;
            // A segment
            const int ra  = p.tA ? (g >> 4) : (g >> 3);
            const int sa4 = (p.tA ? (g & 15) : (g & 7)) * 4;
            const float* ga = p.tA ? (Ab + (size_t)(k0 + ra) * p.lda + rowBase + sa4)
                                   : (Ab + (size_t)(rowBase + ra) * p.lda + k0 + sa4);
            float* la = AsBuf + ra * (p.tA ? 68 : 36) + sa4;
            // B segment
            const int rb  = p.tB ? (g >> 3) : (g >> 4);
            const int sb4 = (p.tB ? (g & 7) : (g & 15)) * 4;
            const float* gb = p.tB ? (Bb + (size_t)(colBase + rb) * p.ldb + k0 + sb4)
                                   : (Bb + (size_t)(k0 + rb) * p.ldb + colBase + sb4);
            float* lb = BsBuf + rb * (p.tB ? 36 : 68) + sb4;
#if USE_ASYNC_LDS
            __builtin_amdgcn_global_load_async_to_lds_b128(
                (g_vi4*)ga, (l_vi4*)la, 0, 0);
            __builtin_amdgcn_global_load_async_to_lds_b128(
                (g_vi4*)gb, (l_vi4*)lb, 0, 0);
#else
            *(float4*)la = *(const float4*)ga;
            *(float4*)lb = *(const float4*)gb;
#endif
            // prefetch next K-chunk (lowers to global_prefetch_b8)
            if (k0 + 32 < p.K) {
                __builtin_prefetch(p.tA ? (ga + (size_t)32 * p.lda) : (ga + 32), 0, 1);
                __builtin_prefetch(p.tB ? (gb + 32) : (gb + (size_t)32 * p.ldb), 0, 1);
            }
        }
#if USE_ASYNC_LDS
        __builtin_amdgcn_s_wait_asynccnt(0);
#endif
        __syncthreads();

        // ---- build f16 fragments per documented CDNA5 16-bit layouts ----
        v16h af[2], bf[2];
        #pragma unroll
        for (int i = 0; i < 2; ++i) {
            const int rA = wr + i * 16 + mloc;
            const int cB = wc + i * 16 + mloc;
            if (!p.tA) {
                const float* ar = AsBuf + (size_t)rA * 36;
                #pragma unroll
                for (int e = 0; e < 8; ++e) {
                    af[i][e]     = (f16)ar[hsel * 8 + e];        // K = hsel*8+e
                    af[i][8 + e] = (f16)ar[16 + hsel * 8 + e];   // K = 16+hsel*8+e
                }
            } else {
                #pragma unroll
                for (int e = 0; e < 8; ++e) {
                    af[i][e]     = (f16)AsBuf[(hsel * 8 + e) * 68 + rA];
                    af[i][8 + e] = (f16)AsBuf[(16 + hsel * 8 + e) * 68 + rA];
                }
            }
            if (!p.tB) {
                #pragma unroll
                for (int e = 0; e < 8; ++e) {
                    bf[i][e]     = (f16)BsBuf[(hsel * 8 + e) * 68 + cB];
                    bf[i][8 + e] = (f16)BsBuf[(16 + hsel * 8 + e) * 68 + cB];
                }
            } else {
                const float* br = BsBuf + (size_t)cB * 36;
                #pragma unroll
                for (int e = 0; e < 8; ++e) {
                    bf[i][e]     = (f16)br[hsel * 8 + e];
                    bf[i][8 + e] = (f16)br[16 + hsel * 8 + e];
                }
            }
        }
        #pragma unroll
        for (int i = 0; i < 2; ++i)
            #pragma unroll
            for (int j = 0; j < 2; ++j)
                acc[i][j] = __builtin_amdgcn_wmma_f32_16x16x32_f16(
                    false, af[i], false, bf[j], (short)0, acc[i][j], false, false);
        __syncthreads();
    }

    // ---- epilogue: C layout lane=N%16, VGPR v -> M = v + 8*half ----
    #pragma unroll
    for (int i = 0; i < 2; ++i) {
        #pragma unroll
        for (int j = 0; j < 2; ++j) {
            const int col = colBase + wc + j * 16 + mloc;
            const float bvv = p.bias ? p.bias[col] : 0.f;
            #pragma unroll
            for (int v = 0; v < 8; ++v) {
                const int row = rowBase + wr + i * 16 + v + 8 * hsel;
                const size_t o = (size_t)row * p.ldc + col;
                const float val = acc[i][j][v] + bvv;
                if (p.accum) Cb[o] += val; else Cb[o] = val;
            }
        }
    }
}

static inline void launch_gemm(hipStream_t s, int batch, int M, int N, int K,
                               const float* A, int lda, long long sA, int tA,
                               const float* B, int ldb, long long sB, int tB,
                               float* C, int ldc, long long sC,
                               const float* bias, int accum) {
    GemmP p;
    p.A = A; p.B = B; p.C = C; p.bias = bias;
    p.M = M; p.N = N; p.K = K; p.lda = lda; p.ldb = ldb; p.ldc = ldc;
    p.sA = sA; p.sB = sB; p.sC = sC; p.tA = tA; p.tB = tB; p.accum = accum;
    dim3 g((unsigned)(N / 64), (unsigned)(M / 64), (unsigned)batch);
    wmma_gemm_kernel<<<g, dim3(128), 0, s>>>(p);
}

// =====================================================================
// Conv stem:  conv1d(k=3, pad=1) + folded-BN affine + ReLU
// =====================================================================
__global__ __launch_bounds__(128) void conv1_kernel(
        const float* x, const float* wgt, const float* bias,
        const float* scale, const float* shift, float* out) {
    const int bt = blockIdx.x;
    const int b = bt / TT, t = bt - b * TT;
    __shared__ float xs[3 * NMEL];
    for (int i = threadIdx.x; i < 3 * NMEL; i += 128) {
        const int dk = i / NMEL, ci = i - dk * NMEL;
        const int tt = t + dk - 1;
        xs[i] = (tt >= 0 && tt < TT) ? x[((size_t)b * TT + tt) * NMEL + ci] : 0.f;
    }
    __syncthreads();
    const int co = threadIdx.x;                // 128 out channels
    float acc = bias[co];
    const float* wr = wgt + (size_t)co * NMEL * 3;
    for (int ci = 0; ci < NMEL; ++ci)
        #pragma unroll
        for (int dk = 0; dk < 3; ++dk)
            acc += xs[dk * NMEL + ci] * wr[ci * 3 + dk];
    const float v = acc * scale[co] + shift[co];
    out[((size_t)b * TT + t) * 128 + co] = fmaxf(v, 0.f);
}

__global__ __launch_bounds__(256) void conv2_kernel(
        const float* x, const float* wgt, const float* bias,
        const float* scale, const float* shift, float* out) {
    const int bt = blockIdx.x;
    const int b = bt / TT, t = bt - b * TT;
    __shared__ float xs[3 * 128];
    for (int i = threadIdx.x; i < 3 * 128; i += 256) {
        const int dk = i / 128, ci = i - dk * 128;
        const int tt = t + dk - 1;
        xs[i] = (tt >= 0 && tt < TT) ? x[((size_t)b * TT + tt) * 128 + ci] : 0.f;
    }
    __syncthreads();
    const int co = threadIdx.x;                // 256 out channels
    float acc = bias[co];
    const float* wr = wgt + (size_t)co * 128 * 3;
    for (int ci = 0; ci < 128; ++ci)
        #pragma unroll
        for (int dk = 0; dk < 3; ++dk)
            acc += xs[dk * 128 + ci] * wr[ci * 3 + dk];
    const float v = acc * scale[co] + shift[co];
    out[((size_t)b * TT + t) * DD + co] = fmaxf(v, 0.f);
}

// =====================================================================
// LayerNorm over last dim (width 256 or 128), optional fused ReLU
// =====================================================================
__global__ __launch_bounds__(128) void layernorm_kernel(
        const float* in, const float* g, const float* bvec, float* out,
        int width, int dorelu) {
    const int row = blockIdx.x;
    const float* p = in + (size_t)row * width;
    float a = 0.f, b2 = 0.f;
    for (int i = threadIdx.x; i < width; i += 128) { float v = p[i]; a += v; b2 += v * v; }
    __shared__ float s1[128], s2[128];
    s1[threadIdx.x] = a; s2[threadIdx.x] = b2; __syncthreads();
    for (int s = 64; s > 0; s >>= 1) {
        if (threadIdx.x < s) { s1[threadIdx.x] += s1[threadIdx.x + s]; s2[threadIdx.x] += s2[threadIdx.x + s]; }
        __syncthreads();
    }
    const float mean = s1[0] / width;
    const float var  = s2[0] / width - mean * mean;
    const float rstd = rsqrtf(var + 1e-5f);
    for (int i = threadIdx.x; i < width; i += 128) {
        float v = (p[i] - mean) * rstd * g[i] + bvec[i];
        out[(size_t)row * width + i] = dorelu ? fmaxf(v, 0.f) : v;
    }
}

// =====================================================================
// Elementwise helpers
// =====================================================================
__global__ void copy_kernel(float* dst, const float* src, long long n) {
    long long i = (long long)blockIdx.x * blockDim.x + threadIdx.x;
    if (i < n) dst[i] = src[i];
}
__global__ void avg_kernel(float* dst, const float* a, const float* b, long long n) {
    long long i = (long long)blockIdx.x * blockDim.x + threadIdx.x;
    if (i < n) dst[i] = 0.5f * (a[i] + b[i]);
}
// [B,T,H*DH] -> [B,H,T,DH] with optional scale (dn for q/k)
__global__ void split_heads_kernel(const float* in, float* out, float scale) {
    long long i = (long long)blockIdx.x * blockDim.x + threadIdx.x;
    if (i >= (long long)BT * DD) return;
    const int d = (int)(i & (DHH - 1));
    const int h = (int)((i >> 6) & (HH - 1));
    const int t = (int)((i >> 8) & (TT - 1));
    const int b = (int)(i >> 19);
    out[((((size_t)b * HH + h) * TT + t) * DHH) + d] = in[i] * scale;
}
// [B,H,T,DH] -> [B,T,H*DH]
__global__ void merge_heads_kernel(const float* in, float* out) {
    long long i = (long long)blockIdx.x * blockDim.x + threadIdx.x;
    if (i >= (long long)BT * DD) return;
    const int d = (int)(i & (DHH - 1));
    const int h = (int)((i >> 6) & (HH - 1));
    const int t = (int)((i >> 8) & (TT - 1));
    const int b = (int)(i >> 19);
    out[i] = in[((((size_t)b * HH + h) * TT + t) * DHH) + d];
}

// =====================================================================
// FAVOR+ pieces
// =====================================================================
__global__ __launch_bounds__(128) void rowmax_kernel(const float* in, float* out, int width) {
    const int row = blockIdx.x;
    const float* p = in + (size_t)row * width;
    float m = -3.4e38f;
    for (int i = threadIdx.x; i < width; i += 128) m = fmaxf(m, p[i]);
    __shared__ float sm[128];
    sm[threadIdx.x] = m; __syncthreads();
    for (int s = 64; s > 0; s >>= 1) {
        if (threadIdx.x < s) sm[threadIdx.x] = fmaxf(sm[threadIdx.x], sm[threadIdx.x + s]);
        __syncthreads();
    }
    if (threadIdx.x == 0) out[row] = sm[0];
}
// out[row] = 0.5 * sum(in[row][:]^2)   (in already scaled by dn)
__global__ void rowsumsq_kernel(const float* in, float* out, int width, int rows) {
    int i = blockIdx.x * blockDim.x + threadIdx.x;
    if (i >= rows) return;
    const float* p = in + (size_t)i * width;
    float a = 0.f;
    for (int c = 0; c < width; ++c) a += p[c] * p[c];
    out[i] = 0.5f * a;
}
__global__ __launch_bounds__(256) void blockmax_kernel(const float* in, long long n, float* part) {
    __shared__ float sm[256];
    const long long stride = (long long)gridDim.x * blockDim.x;
    float m = -3.4e38f;
    for (long long i = (long long)blockIdx.x * blockDim.x + threadIdx.x; i < n; i += stride)
        m = fmaxf(m, in[i]);
    sm[threadIdx.x] = m; __syncthreads();
    for (int s = 128; s > 0; s >>= 1) {
        if (threadIdx.x < s) sm[threadIdx.x] = fmaxf(sm[threadIdx.x], sm[threadIdx.x + s]);
        __syncthreads();
    }
    if (threadIdx.x == 0) part[blockIdx.x] = sm[0];
}
// dd <- ratio * (exp(dd - diag[row] - m) + 1e-4);  m per-row (q) or scalar (k)
__global__ void favor_exp_kernel(float* dd, const float* diag, const float* rmax,
                                 const float* smax, long long n, int width, float ratio) {
    long long i = (long long)blockIdx.x * blockDim.x + threadIdx.x;
    if (i >= n) return;
    const long long row = i / width;
    const float m = rmax ? rmax[row] : smax[0];
    dd[i] = ratio * (expf(dd[i] - diag[row] - m) + 1e-4f);
}
// ksum[bh][m] = sum_n kp[bh][n][m]
__global__ void ksum_kernel(const float* kp, float* ks) {
    int i = blockIdx.x * blockDim.x + threadIdx.x;
    if (i >= BB * HH * MFF) return;
    const int bh = i / MFF, m = i - bh * MFF;
    const float* p = kp + (size_t)bh * TT * MFF + m;
    float a = 0.f;
    for (int n = 0; n < TT; ++n) a += p[(size_t)n * MFF];
    ks[i] = a;
}
// d = qp[bh][n][:] . ksum[bh][:];  oh[bh][n][:] *= 1/d
__global__ void dinv_scale_kernel(const float* qp, const float* ks, float* oh) {
    int i = blockIdx.x * blockDim.x + threadIdx.x;
    if (i >= BHT) return;
    const int bh = i / TT;
    const float* q = qp + (size_t)i * MFF;
    const float* k = ks + (size_t)bh * MFF;
    float d = 0.f;
    for (int m = 0; m < MFF; ++m) d += q[m] * k[m];
    const float inv = 1.f / d;
    float* o = oh + (size_t)i * DHH;
    for (int c = 0; c < DHH; ++c) o[c] *= inv;
}

// GLU with exact GELU:  gact = gelu(ff1[:, :1024]) * ff1[:, 1024:]
__global__ void glu_kernel(const float* ff1, float* gact) {
    long long i = (long long)blockIdx.x * blockDim.x + threadIdx.x;
    if (i >= (long long)BT * 1024) return;
    const long long row = i / 1024;
    const int c = (int)(i - row * 1024);
    const float a = ff1[row * 2048 + c];
    const float g = ff1[row * 2048 + 1024 + c];
    const float ge = 0.5f * a * (1.f + erff(a * 0.70710678118f));
    gact[i] = ge * g;
}

// final head: out[bt] = y[bt][:] . w2[:] + b2
__global__ void clf_final_kernel(const float* y, const float* w2, const float* b2, float* out) {
    int i = blockIdx.x * blockDim.x + threadIdx.x;
    if (i >= BT) return;
    const float* p = y + (size_t)i * 128;
    float acc = b2[0];
    for (int c = 0; c < 128; ++c) acc += p[c] * w2[c];
    out[i] = acc;
}

// =====================================================================
// Host orchestration
// =====================================================================
extern "C" void kernel_launch(void* const* d_in, const int* in_sizes, int n_in,
                              void* d_out, int out_size, void* d_ws, size_t ws_size,
                              hipStream_t stream) {
    (void)in_sizes; (void)n_in; (void)out_size; (void)ws_size;
    const float* x        = (const float*)d_in[0];
    const float* conv1_w  = (const float*)d_in[1];
    const float* conv1_b  = (const float*)d_in[2];
    const float* bn1_s    = (const float*)d_in[3];
    const float* bn1_sh   = (const float*)d_in[4];
    const float* conv2_w  = (const float*)d_in[5];
    const float* conv2_b  = (const float*)d_in[6];
    const float* bn2_s    = (const float*)d_in[7];
    const float* bn2_sh   = (const float*)d_in[8];
    const float* proj_w   = (const float*)d_in[9];
    const float* proj_b   = (const float*)d_in[10];
    const float* norm_g   = (const float*)d_in[11];
    const float* norm_b   = (const float*)d_in[12];
    const float* attn_ln_g= (const float*)d_in[13];
    const float* attn_ln_b= (const float*)d_in[14];
    const float* wq = (const float*)d_in[15]; const float* bq = (const float*)d_in[16];
    const float* wk = (const float*)d_in[17]; const float* bk = (const float*)d_in[18];
    const float* wv = (const float*)d_in[19]; const float* bv = (const float*)d_in[20];
    const float* wo = (const float*)d_in[21]; const float* bo = (const float*)d_in[22];
    const float* proj_mat = (const float*)d_in[23];
    const float* ff_ln_g  = (const float*)d_in[24];
    const float* ff_ln_b  = (const float*)d_in[25];
    const float* ff_w1 = (const float*)d_in[26]; const float* ff_b1 = (const float*)d_in[27];
    const float* ff_w2 = (const float*)d_in[28]; const float* ff_b2 = (const float*)d_in[29];
    const float* clf_w1 = (const float*)d_in[30]; const float* clf_b1 = (const float*)d_in[31];
    const float* clf_ln_g = (const float*)d_in[32]; const float* clf_ln_b = (const float*)d_in[33];
    const float* clf_w2 = (const float*)d_in[34]; const float* clf_b2 = (const float*)d_in[35];
    float* out = (float*)d_out;

    // ---- workspace carve-up (floats) ----
    float* w = (float*)d_ws;
    size_t off = 0;
    auto alloc = [&](size_t n) { float* p = w + off; off += n; return p; };
    const size_t BTD = (size_t)BT * DD;
    float* h1   = alloc((size_t)BT * 128);         // conv1 out (reused: y1)
    float* h2   = alloc(BTD);                      // conv2 out
    float* x1   = alloc(BTD);
    float* x2   = alloc(BTD);
    float* zln  = alloc(BTD);
    float* tmp  = alloc(BTD);                      // q/k/v pre-split, o merged, y1ln
    float* qh   = alloc(BTD);
    float* kh   = alloc(BTD);
    float* vh   = alloc(BTD);
    float* oh   = alloc(BTD);
    float* qp   = alloc((size_t)BHT * MFF);        // reused: ff1 [BT,2048]
    float* kp   = alloc((size_t)BHT * MFF);        // reused: gact [BT,1024]
    float* ctx  = alloc((size_t)BB * HH * MFF * DHH);
    float* rmax = alloc(BHT);
    float* dgb  = alloc(BHT);
    float* part = alloc(1024);
    float* kmx  = alloc(16);
    float* ks   = alloc((size_t)BB * HH * MFF);
    float* ff1  = qp;
    float* gact = kp;
    float* y1   = h1;

    const float dn    = 0.35355339059327373f;   // 64^-0.25
    const float ratio = 0.044194173824159216f;  // 512^-0.5
    const long long sQH = (long long)TT * DHH;  // per-(b,h) stride of qh/kh/vh/oh
    const long long sQP = (long long)TT * MFF;  // per-(b,h) stride of qp/kp
    const long long sCT = (long long)MFF * DHH; // per-(b,h) stride of ctx

    auto ew = [](long long n) { return dim3((unsigned)((n + 255) / 256)); };

    // ---- conv stem ----
    conv1_kernel<<<BT, 128, 0, stream>>>(x, conv1_w, conv1_b, bn1_s, bn1_sh, h1);
    conv2_kernel<<<BT, 256, 0, stream>>>(h1, conv2_w, conv2_b, bn2_s, bn2_sh, h2);

    // ---- input proj + LN, init reversible streams ----
    launch_gemm(stream, 1, BT, DD, DD, h2, DD, 0, 0, proj_w, DD, 0, 0, tmp, DD, 0, proj_b, 0);
    layernorm_kernel<<<BT, 128, 0, stream>>>(tmp, norm_g, norm_b, x1, DD, 0);
    copy_kernel<<<ew(BTD), 256, 0, stream>>>(x2, x1, (long long)BTD);

    for (int i = 0; i < LLAY; ++i) {
        const float* wq_i = wq + (size_t)i * DD * DD;
        const float* wk_i = wk + (size_t)i * DD * DD;
        const float* wv_i = wv + (size_t)i * DD * DD;
        const float* wo_i = wo + (size_t)i * DD * DD;
        const float* pm_i = proj_mat + (size_t)i * MFF * DHH;

        // ----- attention: x1 += Attn(LN(x2)) -----
        layernorm_kernel<<<BT, 128, 0, stream>>>(x2, attn_ln_g + i * DD, attn_ln_b + i * DD, zln, DD, 0);
        launch_gemm(stream, 1, BT, DD, DD, zln, DD, 0, 0, wq_i, DD, 0, 0, tmp, DD, 0, bq + i * DD, 0);
        split_heads_kernel<<<ew(BTD), 256, 0, stream>>>(tmp, qh, dn);
        launch_gemm(stream, 1, BT, DD, DD, zln, DD, 0, 0, wk_i, DD, 0, 0, tmp, DD, 0, bk + i * DD, 0);
        split_heads_kernel<<<ew(BTD), 256, 0, stream>>>(tmp, kh, dn);
        launch_gemm(stream, 1, BT, DD, DD, zln, DD, 0, 0, wv_i, DD, 0, 0, tmp, DD, 0, bv + i * DD, 0);
        split_heads_kernel<<<ew(BTD), 256, 0, stream>>>(tmp, vh, 1.0f);

        // q features: dd = qh @ proj^T  (proj [M,dh] stored [N x K] -> tB=1)
        launch_gemm(stream, BB * HH, TT, MFF, DHH, qh, DHH, sQH, 0, pm_i, DHH, 0, 1, qp, MFF, sQP, nullptr, 0);
        rowmax_kernel<<<BHT, 128, 0, stream>>>(qp, rmax, MFF);
        rowsumsq_kernel<<<ew(BHT), 256, 0, stream>>>(qh, dgb, DHH, BHT);
        favor_exp_kernel<<<ew((long long)BHT * MFF), 256, 0, stream>>>(qp, dgb, rmax, nullptr, (long long)BHT * MFF, MFF, ratio);

        // k features: global max over whole dd tensor
        launch_gemm(stream, BB * HH, TT, MFF, DHH, kh, DHH, sQH, 0, pm_i, DHH, 0, 1, kp, MFF, sQP, nullptr, 0);
        rowsumsq_kernel<<<ew(BHT), 256, 0, stream>>>(kh, dgb, DHH, BHT);
        blockmax_kernel<<<1024, 256, 0, stream>>>(kp, (long long)BHT * MFF, part);
        blockmax_kernel<<<1, 256, 0, stream>>>(part, 1024, kmx);
        favor_exp_kernel<<<ew((long long)BHT * MFF), 256, 0, stream>>>(kp, dgb, nullptr, kmx, (long long)BHT * MFF, MFF, ratio);

        // linear attention
        ksum_kernel<<<ew(BB * HH * MFF), 256, 0, stream>>>(kp, ks);
        // ctx = kp^T @ v  (A stored [K=T x M=MFF] -> tA=1)
        launch_gemm(stream, BB * HH, MFF, DHH, TT, kp, MFF, sQP, 1, vh, DHH, sQH, 0, ctx, DHH, sCT, nullptr, 0);
        // o = qp @ ctx
        launch_gemm(stream, BB * HH, TT, DHH, MFF, qp, MFF, sQP, 0, ctx, DHH, sCT, 0, oh, DHH, sQH, nullptr, 0);
        dinv_scale_kernel<<<ew(BHT), 256, 0, stream>>>(qp, ks, oh);
        merge_heads_kernel<<<ew(BTD), 256, 0, stream>>>(oh, tmp);
        // x1 += o @ wo + bo  (residual fused via accumulate epilogue)
        launch_gemm(stream, 1, BT, DD, DD, tmp, DD, 0, 0, wo_i, DD, 0, 0, x1, DD, 0, bo + i * DD, 1);

        // ----- FF: x2 += FF(LN(x1)) -----
        layernorm_kernel<<<BT, 128, 0, stream>>>(x1, ff_ln_g + i * DD, ff_ln_b + i * DD, zln, DD, 0);
        launch_gemm(stream, 1, BT, 8 * DD, DD, zln, DD, 0, 0,
                    ff_w1 + (size_t)i * DD * 8 * DD, 8 * DD, 0, 0, ff1, 8 * DD, 0, ff_b1 + i * 8 * DD, 0);
        glu_kernel<<<ew((long long)BT * 1024), 256, 0, stream>>>(ff1, gact);
        launch_gemm(stream, 1, BT, DD, 4 * DD, gact, 4 * DD, 0, 0,
                    ff_w2 + (size_t)i * 4 * DD * DD, DD, 0, 0, x2, DD, 0, ff_b2 + i * DD, 1);
    }

    // ---- classifier ----
    avg_kernel<<<ew(BTD), 256, 0, stream>>>(zln, x1, x2, (long long)BTD);
    launch_gemm(stream, 1, BT, 128, DD, zln, DD, 0, 0, clf_w1, 128, 0, 0, y1, 128, 0, clf_b1, 0);
    layernorm_kernel<<<BT, 128, 0, stream>>>(y1, clf_ln_g, clf_ln_b, tmp, 128, 1);
    clf_final_kernel<<<ew(BT), 256, 0, stream>>>(tmp, clf_w2, clf_b2, out);
}
